// full_conv_Layer_77369540870267
// MI455X (gfx1250) — matrix operand order repaired and checked by reading, not compiled, not run on hardware
//
#include <hip/hip_runtime.h>
#include <hip/hip_bf16.h>

// Problem constants (from reference)
#define B_  32
#define P_  4096
#define M_  12
#define W_  9
#define C_  64          // IN_C == OUT_C == 64
#define NPAD 4097       // IN_PN_PAD
#define K_  (W_ * C_)   // 576 : GEMM reduction dim
#define PITCH 584       // 576 + 8 f16 pad -> 292 dwords row pitch, conflict-free b128

typedef __attribute__((ext_vector_type(16))) _Float16  v16h;
typedef __attribute__((ext_vector_type(2)))  _Float16  v2h;
typedef __attribute__((ext_vector_type(2)))  __fp16    v2hf;   // builtin return type
typedef __attribute__((ext_vector_type(8)))  float     v8f;
typedef __attribute__((ext_vector_type(4)))  float     v4f;
typedef __attribute__((ext_vector_type(4)))  unsigned  v4u;

union H16 { v16h h; v4u u[2]; };
union HU  { v2h  h; unsigned u; };
union HP  { v2h  h[4]; v4u u; };

// v_cvt_pk_rtz_f16_f32 wrapper with the vector element type we do math in
static __device__ __forceinline__ v2h cvt_pk(float a, float b) {
    v2hf r = __builtin_amdgcn_cvt_pkrtz(a, b);
    return __builtin_bit_cast(v2h, r);
}

// --- prep: WcT[o][k] = (f16) weights[w, o*64+i],  k = w*64+i ;  64 x 576 ---
__global__ void prep_wct_kernel(const float* __restrict__ weights,
                                _Float16* __restrict__ WcT) {
    int idx = blockIdx.x * 256 + threadIdx.x;      // 0 .. 36863
    if (idx >= C_ * K_) return;
    int o = idx / K_;
    int k = idx - o * K_;
    int w = k >> 6;
    int i = k & 63;
    WcT[idx] = (_Float16)weights[w * (C_ * C_) + o * C_ + i];
}

// --- main: one block per point p ---
__global__ __launch_bounds__(256)
void full_conv_main_kernel(const float* __restrict__ x,     // (32,4097,64) f32
                           const float* __restrict__ bias,  // (64)
                           const float* __restrict__ ww,    // (P,12,9)
                           const int*   __restrict__ nid,   // (P,12)
                           const float* __restrict__ mask,  // (P,12)
                           const _Float16* __restrict__ WcT,// (64,576) f16
                           float* __restrict__ out)         // (32,P,64)
{
    __shared__ _Float16 Ylds[B_ * PITCH];   // 36.5 KB: Y[b][k] in f16
    __shared__ unsigned w2h[M_ * W_];       // 108 combined weights as broadcast half2
    __shared__ int      nids[M_];

    const int p   = blockIdx.x;
    const int tid = threadIdx.x;

    if (tid < M_ * W_) {
        float v = ww[p * (M_ * W_) + tid] * mask[p * M_ + tid / W_];
        HU t; t.h = cvt_pk(v, v);                    // duplicate into both halves
        w2h[tid] = t.u;
    } else if (tid < M_ * W_ + M_) {
        nids[tid - M_ * W_] = nid[p * M_ + (tid - M_ * W_)];
    }
    __syncthreads();

    // ---- Phase 1: Y[b, w*64+i] = sum_m w2[m,w] * x[b, nid[m], i]  (packed f16 VALU) ----
    {
        const int b  = tid >> 3;          // 0..31
        const int ic = (tid & 7) * 8;     // 0,8,...,56  (8 channels per thread)

        v2h acc[W_][4];
        #pragma unroll
        for (int w = 0; w < W_; ++w)
            #pragma unroll
            for (int j = 0; j < 4; ++j) acc[w][j] = (v2h)((_Float16)0.0f);

        #pragma unroll
        for (int m = 0; m < M_; ++m) {
            const float* xp = x + ((size_t)b * NPAD + nids[m]) * C_ + ic;
            v4f x0 = *reinterpret_cast<const v4f*>(xp);
            v4f x1 = *reinterpret_cast<const v4f*>(xp + 4);
            v2h xh[4];
            xh[0] = cvt_pk(x0[0], x0[1]);            // v_cvt_pk_rtz_f16_f32
            xh[1] = cvt_pk(x0[2], x0[3]);
            xh[2] = cvt_pk(x1[0], x1[1]);
            xh[3] = cvt_pk(x1[2], x1[3]);
            #pragma unroll
            for (int w = 0; w < W_; ++w) {
                HU t; t.u = w2h[m * W_ + w];
                #pragma unroll
                for (int j = 0; j < 4; ++j)
                    acc[w][j] += xh[j] * t.h;        // v_pk_fma_f16
            }
        }
        #pragma unroll
        for (int w = 0; w < W_; ++w) {
            HP pk;
            #pragma unroll
            for (int j = 0; j < 4; ++j) pk.h[j] = acc[w][j];
            *reinterpret_cast<v4u*>(&Ylds[b * PITCH + w * C_ + ic]) = pk.u;  // ds_store_b128
        }
    }
    __syncthreads();

    // ---- Phase 2: out(32x64) = Y(32x576) x WcT^T via v_wmma_f32_16x16x32_f16 ----
    const int wave = tid >> 5;          // 0..7
    const int lane = tid & 31;
    const int mt   = wave >> 2;         // row tile 0..1  (b in [mt*16, mt*16+16))
    const int nt   = wave & 3;          // col tile 0..3  (o in [nt*16, nt*16+16))
    const int lrow = lane & 15;
    const bool hi  = (lane >= 16);

    // A (16x32 f16) ISA layout: lane<16: halves = K 0..7 | 16..23 ; lane>=16: K 8..15 | 24..31
    const int aBase = (mt * 16 + lrow) * PITCH + (hi ? 8 : 0);
    // B (32x16 f16) layout matching A: col = lane%16 ; lane<16: K 0..7 | 16..23 ; lane>=16: K 8..15 | 24..31
    const _Float16* bp = WcT + (nt * 16 + lrow) * K_ + (hi ? 8 : 0);

    v8f acc = {0.f, 0.f, 0.f, 0.f, 0.f, 0.f, 0.f, 0.f};

    #pragma unroll
    for (int kk = 0; kk < K_ / 32; ++kk) {          // 18 K-steps
        H16 A, Bm;
        A.u[0]  = *reinterpret_cast<const v4u*>(&Ylds[aBase + kk * 32]);       // K .. K+7
        A.u[1]  = *reinterpret_cast<const v4u*>(&Ylds[aBase + kk * 32 + 16]);  // K+16 .. K+23
        Bm.u[0] = *reinterpret_cast<const v4u*>(bp + kk * 32);                 // K .. K+7
        Bm.u[1] = *reinterpret_cast<const v4u*>(bp + kk * 32 + 16);
        acc = __builtin_amdgcn_wmma_f32_16x16x32_f16(
                  /*neg_a=*/false, A.h, /*neg_b=*/false, Bm.h,
                  /*c_mod=*/(short)0, acc, /*reuse_a=*/false, /*reuse_b=*/false);
    }

    // C/D layout: VGPR r -> row (r or r+8), col = lane%16
    const int   o     = nt * 16 + lrow;
    const float bv    = bias[o];
    const int   brow0 = mt * 16 + (hi ? 8 : 0);
    #pragma unroll
    for (int r = 0; r < 8; ++r) {
        out[((size_t)(brow0 + r) * P_ + p) * C_ + o] = acc[r] + bv;
    }
}

extern "C" void kernel_launch(void* const* d_in, const int* in_sizes, int n_in,
                              void* d_out, int out_size, void* d_ws, size_t ws_size,
                              hipStream_t stream) {
    const float* x       = (const float*)d_in[0];   // in_pc_pad (32,4097,64)
    const float* weights = (const float*)d_in[1];   // (9, 4096)
    const float* bias    = (const float*)d_in[2];   // (64)
    const float* ww      = (const float*)d_in[3];   // w_weights (4096,12,9)
    const int*   nid     = (const int*)d_in[4];     // neighbor ids (4096,12)
    const float* mask    = (const float*)d_in[5];   // (4096,12)

    _Float16* WcT = (_Float16*)d_ws;                // 64*576 f16 = 72 KB scratch

    prep_wct_kernel<<<(C_ * K_ + 255) / 256, 256, 0, stream>>>(weights, WcT);
    full_conv_main_kernel<<<P_, 256, 0, stream>>>(x, bias, ww, nid, mask, WcT,
                                                  (float*)d_out);
}
